// DIMKT_14946486190534
// MI455X (gfx1250) — matrix operand order repaired and checked by reading, not compiled, not run on hardware
//
#include <hip/hip_runtime.h>

typedef __attribute__((ext_vector_type(16))) __bf16 v16bf;
typedef __attribute__((ext_vector_type(8)))  float  v8f;

union FragU { v16bf v; unsigned u[8]; };

__device__ __forceinline__ unsigned pack2bf(float a, float b) {
  union { __bf16 h[2]; unsigned u; } p;
  p.h[0] = (__bf16)a; p.h[1] = (__bf16)b;
  return p.u;
}
__device__ __forceinline__ float sigmf(float x) { return 1.0f / (1.0f + __expf(-x)); }

#define Bsz 1024
#define Ssz 200
#define Dsz 128

// ---------------------------------------------------------------------------
// Kernel P: fold ct/qdt/cdt contributions of Wk/Wp1/Wp2 into lookup tables.
//   Qk3[q,n] = Eqd[q] . Wk[n, 256:384]
//   Ck4[c,n] = Ecd[c] . Wk[n, 384:512]
//   Ck2[j,n] = Ecorr[j] . Wk[n, 128:256] + bk[n]
//   Cp1[j,n] = Ecorr[j] . Wp1[n, 128:256] + bp1[n]
//   Cp2[j,n] = Ecorr[j] . Wp2[n, 128:256] + bp2[n]
// ---------------------------------------------------------------------------
__global__ __launch_bounds__(256) void dimkt_tables(
    const float* __restrict__ Eqd, const float* __restrict__ Ecd,
    const float* __restrict__ Ecorr,
    const float* __restrict__ Wp1, const float* __restrict__ bp1,
    const float* __restrict__ Wp2, const float* __restrict__ bp2,
    const float* __restrict__ Wk,  const float* __restrict__ bk,
    float* __restrict__ Qk3, float* __restrict__ Ck4, float* __restrict__ Ck2,
    float* __restrict__ Cp1, float* __restrict__ Cp2) {
  int gid = blockIdx.x * blockDim.x + threadIdx.x;
  if (gid >= 208 * 128) return;
  int r = gid >> 7, n = gid & 127;
  float s = 0.f;
  if (r < 101) {
    const float* e = Eqd + r * 128; const float* w = Wk + n * 512 + 256;
    for (int k = 0; k < 128; ++k) s += e[k] * w[k];
    Qk3[r * 128 + n] = s;
  } else if (r < 202) {
    const float* e = Ecd + (r - 101) * 128; const float* w = Wk + n * 512 + 384;
    for (int k = 0; k < 128; ++k) s += e[k] * w[k];
    Ck4[(r - 101) * 128 + n] = s;
  } else if (r < 204) {
    const float* e = Ecorr + (r - 202) * 128; const float* w = Wk + n * 512 + 128;
    for (int k = 0; k < 128; ++k) s += e[k] * w[k];
    Ck2[(r - 202) * 128 + n] = s + bk[n];
  } else if (r < 206) {
    const float* e = Ecorr + (r - 204) * 128; const float* w = Wp1 + n * 256 + 128;
    for (int k = 0; k < 128; ++k) s += e[k] * w[k];
    Cp1[(r - 204) * 128 + n] = s + bp1[n];
  } else {
    const float* e = Ecorr + (r - 206) * 128; const float* w = Wp2 + n * 256 + 128;
    for (int k = 0; k < 128; ++k) s += e[k] * w[k];
    Cp2[(r - 206) * 128 + n] = s + bp2[n];
  }
}

// ---------------------------------------------------------------------------
// Kernel X: x = concat(qe,ce,qde,cde) @ Wx^T + bx  ([B*S,512] x [512,128])
// Per block: Wx cached in LDS (bf16, pair-packed) once; 8 row-tiles of 16 rows.
// bf16 WMMA 16x16x32, f32 accumulate.
// ---------------------------------------------------------------------------
__global__ __launch_bounds__(256) void dimkt_xmlp(
    const int* __restrict__ qseq, const int* __restrict__ cseq,
    const int* __restrict__ qdseq, const int* __restrict__ cdseq,
    const float* __restrict__ Eq, const float* __restrict__ Ec,
    const float* __restrict__ Eqd, const float* __restrict__ Ecd,
    const float* __restrict__ Wx, const float* __restrict__ bx,
    float* __restrict__ xout) {
  extern __shared__ char smem[];
  __bf16* Bh = (__bf16*)smem;      // Wx bf16 pair-packed: idx = (k>>1)*256 + n*2 + (k&1)
  __bf16* Ah = Bh + 512 * 128;     // A tile [16][520] bf16 (row-padded)
  const unsigned* Bu = (const unsigned*)Bh;

  int tid = threadIdx.x;
  int wave = tid >> 5, lane = tid & 31, hi = lane >> 4, l15 = lane & 15;
  int nc = wave * 16 + l15;        // this lane's output / B column

  // Load Wx (f32 row-major [128][512]) -> LDS bf16 pair-packed (B layout fast path)
  for (int idx = tid; idx < 512 * 128; idx += 256) {
    int n = idx >> 9, k = idx & 511;
    Bh[(k >> 1) * 256 + n * 2 + (k & 1)] = (__bf16)Wx[idx];
  }
  float bxv = bx[nc];

  for (int rt = 0; rt < 8; ++rt) {
    int row0 = blockIdx.x * 128 + rt * 16;
    __syncthreads();  // Bh visible; previous tile's Ah reads done
    // Gather 16 rows x 512 concat-embedding into Ah (bf16)
    for (int idx = tid; idx < 16 * 512; idx += 256) {
      int r = idx >> 9, c = idx & 511;
      int g = row0 + r;
      int b = g / Ssz, s2 = g - b * Ssz;
      int e = c >> 7, j = c & 127;
      float val;
      if (e == 0)      val = Eq [qseq [b * Ssz + s2] * 128 + j];
      else if (e == 1) val = Ec [cseq [b * Ssz + s2] * 128 + j];
      else if (e == 2) val = Eqd[qdseq[b * Ssz + s2] * 128 + j];
      else             val = Ecd[cdseq[b * Ssz + s2] * 128 + j];
      Ah[r * 520 + c] = (__bf16)val;
    }
    __syncthreads();

    v8f acc = {};
    #pragma unroll
    for (int kk = 0; kk < 16; ++kk) {
      FragU af, bfr;
      #pragma unroll
      for (int v = 0; v < 8; ++v) {
        int ka = kk * 32 + (v >> 2) * 16 + hi * 8 + (v & 3) * 2;  // A-layout pair
        af.u[v]  = *(const unsigned*)(Ah + l15 * 520 + ka);
        bfr.u[v] = Bu[(kk * 16 + (v >> 2) * 8 + hi * 4 + (v & 3)) * 128 + nc];         // B-layout pair
      }
      acc = __builtin_amdgcn_wmma_f32_16x16x32_bf16(false, af.v, false, bfr.v,
                                                    (short)0, acc, false, false);
    }
    #pragma unroll
    for (int i = 0; i < 8; ++i) {
      int m = i + 8 * hi;                    // C layout: row = i + 8*hi, col = nc
      xout[(row0 + m) * 128 + nc] = acc[i] + bxv;
    }
  }
}

// ---------------------------------------------------------------------------
// Kernel S: persistent recurrent scan. 64 blocks x 16 batch rows x 199 steps.
// 8 waves per block, each owns a 16-column slice of N=128.
// Folded weights (5 x 128x128) live as bf16 B-fragments in VGPRs (160/lane).
// h / x(double-buffered) / sdf live in LDS. 20 bf16 WMMAs per wave per step.
// ---------------------------------------------------------------------------
__global__ __launch_bounds__(256) void dimkt_scan(
    const int* __restrict__ qdseq, const int* __restrict__ cdseq,
    const int* __restrict__ crseq,
    const float* __restrict__ Ws1, const float* __restrict__ bs1,
    const float* __restrict__ Ws2, const float* __restrict__ bs2,
    const float* __restrict__ Wp1, const float* __restrict__ Wp2,
    const float* __restrict__ Wk,
    const float* __restrict__ h0, const float* __restrict__ xws,
    const float* __restrict__ Qk3, const float* __restrict__ Ck4,
    const float* __restrict__ Ck2, const float* __restrict__ Cp1,
    const float* __restrict__ Cp2,
    float* __restrict__ yout) {
  extern __shared__ char smem[];
  float* hb    = (float*)smem;            // [16][132] hidden state
  float* xb    = hb + 16 * 132;           // [2][16][132] x_t / x_{t+1}
  float* sdfb  = xb + 2 * 16 * 132;       // [16][132]
  int*   qdA   = (int*)(sdfb + 16 * 132); // [16]
  int*   cdA   = qdA + 16;                // [16]
  int*   crA   = cdA + 16;                // [16]
  float* ypart = (float*)(crA + 16);      // [16][8]

  int tid = threadIdx.x;
  int wave = tid >> 5, lane = tid & 31, hi = lane >> 4, l15 = lane & 15;
  int nc = wave * 16 + l15;
  int b0 = blockIdx.x * 16;

  // ---- B-fragments of the five folded weights, kept in registers ----------
  v16bf wS1f[4], wS2f[4], wP1f[4], wP2f[4], wKf[4];
  {
    FragU t;
    #pragma unroll
    for (int kk = 0; kk < 4; ++kk) {
      #pragma unroll
      for (int v = 0; v < 8; ++v) {
        int k = kk * 32 + hi * 16 + 2 * v;
        t.u[v] = pack2bf(Ws1[nc * 128 + k], Ws1[nc * 128 + k + 1]);
      }
      wS1f[kk] = t.v;
      #pragma unroll
      for (int v = 0; v < 8; ++v) {
        int k = kk * 32 + hi * 16 + 2 * v;
        t.u[v] = pack2bf(Ws2[nc * 128 + k], Ws2[nc * 128 + k + 1]);
      }
      wS2f[kk] = t.v;
      #pragma unroll
      for (int v = 0; v < 8; ++v) {
        int k = kk * 32 + hi * 16 + 2 * v;
        t.u[v] = pack2bf(Wp1[nc * 256 + k], Wp1[nc * 256 + k + 1]);  // sdf half
      }
      wP1f[kk] = t.v;
      #pragma unroll
      for (int v = 0; v < 8; ++v) {
        int k = kk * 32 + hi * 16 + 2 * v;
        t.u[v] = pack2bf(Wp2[nc * 256 + k], Wp2[nc * 256 + k + 1]);
      }
      wP2f[kk] = t.v;
      #pragma unroll
      for (int v = 0; v < 8; ++v) {
        int k = kk * 32 + hi * 16 + 2 * v;
        t.u[v] = pack2bf(Wk[nc * 512 + k], Wk[nc * 512 + k + 1]);    // h half
      }
      wKf[kk] = t.v;
    }
  }
  float bs1v = bs1[nc], bs2v = bs2[nc];
  float cp1v0 = Cp1[nc], cp1v1 = Cp1[128 + nc];
  float cp2v0 = Cp2[nc], cp2v1 = Cp2[128 + nc];
  float ck2v0 = Ck2[nc], ck2v1 = Ck2[128 + nc];

  // ---- init h and x_0 ------------------------------------------------------
  for (int idx = tid; idx < 2048; idx += 256) {
    int r = idx >> 7, c = idx & 127;
    hb[r * 132 + c] = h0[(b0 + r) * 128 + c];
    xb[r * 132 + c] = xws[((b0 + r) * Ssz + 0) * 128 + c];
  }
  __syncthreads();

  for (int t = 0; t < Ssz - 1; ++t) {
    int cur = t & 1, nxt = cur ^ 1;
    float* xc = xb + cur * 2112;
    float* xn = xb + nxt * 2112;

    // stage x_{t+1} and per-step indices (visible after barrier B1)
    for (int idx = tid; idx < 2048; idx += 256) {
      int r = idx >> 7, c = idx & 127;
      xn[r * 132 + c] = xws[((b0 + r) * Ssz + (t + 1)) * 128 + c];
    }
    if (tid < 16)      qdA[tid]      = qdseq[(b0 + tid) * Ssz + t];
    else if (tid < 32) cdA[tid - 16] = cdseq[(b0 + tid - 16) * Ssz + t];
    else if (tid < 48) crA[tid - 32] = crseq[(b0 + tid - 32) * Ssz + t];

    // ---- phase 1: sdf = sigmoid(s_in Ws1^T + b) * tanh(s_in Ws2^T + b) ----
    v8f a1 = {}, a2 = {};
    #pragma unroll
    for (int kk = 0; kk < 4; ++kk) {
      FragU af;
      #pragma unroll
      for (int v = 0; v < 8; ++v) {
        int k = kk * 32 + hi * 16 + 2 * v;
        float f0 = xc[l15 * 132 + k]     - hb[l15 * 132 + k];
        float f1 = xc[l15 * 132 + k + 1] - hb[l15 * 132 + k + 1];
        af.u[v] = pack2bf(f0, f1);
      }
      a1 = __builtin_amdgcn_wmma_f32_16x16x32_bf16(false, af.v, false, wS1f[kk], (short)0, a1, false, false);
      a2 = __builtin_amdgcn_wmma_f32_16x16x32_bf16(false, af.v, false, wS2f[kk], (short)0, a2, false, false);
    }
    #pragma unroll
    for (int i = 0; i < 8; ++i) {
      int m = i + 8 * hi;
      sdfb[m * 132 + nc] = sigmf(a1[i] + bs1v) * tanhf(a2[i] + bs2v);
    }
    __syncthreads();  // B1: sdf, x_{t+1}, indices visible

    // ---- phase 2: pka, gate g, h update -----------------------------------
    v8f p1 = {}, p2 = {}, gg = {};
    #pragma unroll
    for (int kk = 0; kk < 4; ++kk) {
      FragU as, ah;
      #pragma unroll
      for (int v = 0; v < 8; ++v) {
        int k = kk * 32 + hi * 16 + 2 * v;
        as.u[v] = pack2bf(sdfb[l15 * 132 + k], sdfb[l15 * 132 + k + 1]);
        ah.u[v] = pack2bf(hb[l15 * 132 + k],   hb[l15 * 132 + k + 1]);
      }
      p1 = __builtin_amdgcn_wmma_f32_16x16x32_bf16(false, as.v, false, wP1f[kk], (short)0, p1, false, false);
      p2 = __builtin_amdgcn_wmma_f32_16x16x32_bf16(false, as.v, false, wP2f[kk], (short)0, p2, false, false);
      gg = __builtin_amdgcn_wmma_f32_16x16x32_bf16(false, ah.v, false, wKf[kk],  (short)0, gg, false, false);
    }
    float hn[8];
    #pragma unroll
    for (int i = 0; i < 8; ++i) {
      int m = i + 8 * hi;
      int cr = crA[m];
      float pka = sigmf(p1[i] + (cr ? cp1v1 : cp1v0)) * tanhf(p2[i] + (cr ? cp2v1 : cp2v0));
      float gp  = gg[i] + Qk3[qdA[m] * 128 + nc] + Ck4[cdA[m] * 128 + nc]
                        + (cr ? ck2v1 : ck2v0);
      float g = sigmf(gp);
      float hold = hb[m * 132 + nc];
      hn[i] = g * hold + (1.f - g) * pka;
    }
    __syncthreads();  // B2: all reads of h_pre complete

    // write h_new; y partials: sigmoid(dot(x_{t+1}, h_new)) ------------------
    float yl[8];
    #pragma unroll
    for (int i = 0; i < 8; ++i) {
      int m = i + 8 * hi;
      hb[m * 132 + nc] = hn[i];
      float v = hn[i] * xn[m * 132 + nc];
      v += __shfl_xor(v, 1);
      v += __shfl_xor(v, 2);
      v += __shfl_xor(v, 4);
      v += __shfl_xor(v, 8);   // butterfly within each 16-lane half
      yl[i] = v;
    }
    if (l15 == 0) {
      #pragma unroll
      for (int i = 0; i < 8; ++i) ypart[(i + 8 * hi) * 8 + wave] = yl[i];
    }
    __syncthreads();  // B3: h_new + ypart visible
    if (tid < 16) {
      float s = 0.f;
      #pragma unroll
      for (int w = 0; w < 8; ++w) s += ypart[tid * 8 + w];
      yout[(b0 + tid) * Ssz + t] = sigmf(s);
    }
  }
  if (tid < 16) yout[(b0 + tid) * Ssz + (Ssz - 1)] = 0.f;
}

// ---------------------------------------------------------------------------
extern "C" void kernel_launch(void* const* d_in, const int* in_sizes, int n_in,
                              void* d_out, int out_size, void* d_ws, size_t ws_size,
                              hipStream_t stream) {
  const int*   qseq  = (const int*)d_in[0];
  const int*   cseq  = (const int*)d_in[1];
  const int*   qdseq = (const int*)d_in[2];
  const int*   cdseq = (const int*)d_in[3];
  const int*   crseq = (const int*)d_in[4];
  const float* Eq    = (const float*)d_in[5];
  const float* Ec    = (const float*)d_in[6];
  const float* Eqd   = (const float*)d_in[7];
  const float* Ecd   = (const float*)d_in[8];
  const float* Ecorr = (const float*)d_in[9];
  const float* Wx    = (const float*)d_in[10];
  const float* bx    = (const float*)d_in[11];
  const float* Ws1   = (const float*)d_in[12];
  const float* bs1   = (const float*)d_in[13];
  const float* Ws2   = (const float*)d_in[14];
  const float* bs2   = (const float*)d_in[15];
  const float* Wp1   = (const float*)d_in[16];
  const float* bp1   = (const float*)d_in[17];
  const float* Wp2   = (const float*)d_in[18];
  const float* bp2   = (const float*)d_in[19];
  const float* Wk    = (const float*)d_in[20];
  const float* bk    = (const float*)d_in[21];
  const float* h0    = (const float*)d_in[22];
  float* yout = (float*)d_out;

  // workspace layout (floats)
  float* xws = (float*)d_ws;                  // [B*S*D] = 26,214,400
  float* Qk3 = xws + (size_t)Bsz * Ssz * Dsz; // [101*128]
  float* Ck4 = Qk3 + 101 * 128;               // [101*128]
  float* Ck2 = Ck4 + 101 * 128;               // [2*128]
  float* Cp1 = Ck2 + 256;                     // [2*128]
  float* Cp2 = Cp1 + 256;                     // [2*128]

  dimkt_tables<<<104, 256, 0, stream>>>(Eqd, Ecd, Ecorr, Wp1, bp1, Wp2, bp2,
                                        Wk, bk, Qk3, Ck4, Ck2, Cp1, Cp2);

  size_t smemX = (size_t)(512 * 128 + 16 * 520) * 2;  // 147,712 B
  dimkt_xmlp<<<(Bsz * Ssz) / 128, 256, smemX, stream>>>(
      qseq, cseq, qdseq, cdseq, Eq, Ec, Eqd, Ecd, Wx, bx, xws);

  size_t smemS = (size_t)(16 * 132 + 2 * 16 * 132 + 16 * 132) * 4 + 48 * 4 + 16 * 8 * 4;
  dimkt_scan<<<Bsz / 16, 256, smemS, stream>>>(
      qdseq, cdseq, crseq, Ws1, bs1, Ws2, bs2, Wp1, Wp2, Wk,
      h0, xws, Qk3, Ck4, Ck2, Cp1, Cp2, yout);
}
